// GraphDistanceModel_87677462380867
// MI455X (gfx1250) — matrix operand order repaired and checked by reading, not compiled, not run on hardware
//
#include <hip/hip_runtime.h>

typedef __attribute__((ext_vector_type(16))) __bf16 v16bf;
typedef __attribute__((ext_vector_type(8)))  __bf16 v8bf;
typedef __attribute__((ext_vector_type(8)))  float  v8f;

#define NN 100000
#define NE 1250000
#define NP 200000
#define HID 64
#define IN_DIM 128

// ---- WMMA VGPR layout helpers (wave32, gfx1250) --------------------------
// 16-bit A-matrix 16x32: lane L holds row m=L&15, group g=L>>4.
// vector element i maps to K = (i>>3)*16 + g*8 + (i&7)  (each half = 8 contiguous K)
__device__ __forceinline__ int a_kidx(int i, int g) {
  return ((i >> 3) << 4) + (g << 3) + (i & 7);
}

// assemble v16bf A-operand from a bf16 row (two contiguous 16B halves)
__device__ __forceinline__ v16bf load_a_bf16(const __bf16* row, int kb, int g) {
  v8bf lo = *(const v8bf*)(row + kb + g * 8);
  v8bf hi = *(const v8bf*)(row + kb + 16 + g * 8);
  v16bf a;
#pragma unroll
  for (int i = 0; i < 8; ++i) { a[i] = lo[i]; a[8 + i] = hi[i]; }
  return a;
}

// assemble v16bf A-operand from an f32 row with optional scale
__device__ __forceinline__ v16bf load_a_f32(const float* row, int kb, int g, float s) {
  v16bf a;
#pragma unroll
  for (int i = 0; i < 16; ++i) a[i] = (__bf16)(row[kb + a_kidx(i, g)] * s);
  return a;
}

// read B-tile (col n, k = g*8+i and 16+g*8+i) from column-major bf16 LDS image
__device__ __forceinline__ v16bf load_b_lds(const __bf16* lw, int kdim, int n, int kb, int g) {
  v8bf lo = *(const v8bf*)(lw + (size_t)n * kdim + kb + g * 8);
  v8bf hi = *(const v8bf*)(lw + (size_t)n * kdim + kb + 16 + g * 8);
  v16bf b;
#pragma unroll
  for (int i = 0; i < 8; ++i) { b[i] = lo[i]; b[8 + i] = hi[i]; }
  return b;
}

#define WMMA_BF16(a, b, c) \
  __builtin_amdgcn_wmma_f32_16x16x32_bf16(false, (a), false, (b), (short)0, (c), false, false)

// ---- tiny utility kernels ------------------------------------------------
__global__ void k_zero_f32(float* __restrict__ p, int n) {
  int i = blockIdx.x * blockDim.x + threadIdx.x;
  if (i < n) p[i] = 0.0f;
}

__global__ void k_deg(const int* __restrict__ ei, float* __restrict__ deg, int E) {
  int e = blockIdx.x * blockDim.x + threadIdx.x;
  if (e < E) atomicAdd(&deg[ei[E + e]], 1.0f);   // ei row 1 = dst
}

__global__ void k_invdeg(float* __restrict__ p, int n) {
  int i = blockIdx.x * blockDim.x + threadIdx.x;
  if (i < n) { float d = p[i]; p[i] = (d > 0.0f) ? (1.0f / d) : 0.0f; }
}

// ---- encoder: h = relu(x @ enc_W + enc_b), x f32 [N,128] -> h bf16 [N,64]
// one wave handles M=32 (two 16-row tiles); each B fragment feeds 2 WMMAs
__global__ void k_encoder(const float* __restrict__ x, const float* __restrict__ W,
                          const float* __restrict__ bias, __bf16* __restrict__ h,
                          int n_waves) {
  __shared__ __bf16 lw[HID * IN_DIM];            // column-major: lw[n*128 + k], 16KB
  for (int t = threadIdx.x; t < HID * IN_DIM; t += blockDim.x) {
    int n = t >> 7, k = t & 127;
    lw[t] = (__bf16)W[(size_t)k * HID + n];
  }
  __syncthreads();

  int wave = (int)((blockIdx.x * blockDim.x + threadIdx.x) >> 5);
  int lane = threadIdx.x & 31;
  if (wave >= n_waves) return;                   // wave-uniform: EXEC all-1 at WMMA
  int m = lane & 15, g = lane >> 4;
  int row0 = wave * 32;
  const float* xr0 = x + (size_t)(row0 + m) * IN_DIM;
  const float* xr1 = x + (size_t)(row0 + 16 + m) * IN_DIM;

  v8f acc0[4] = {}, acc1[4] = {};
#pragma unroll
  for (int kc = 0; kc < 4; ++kc) {               // K = 128 in chunks of 32
    int kb = kc * 32;
    v16bf a0 = load_a_f32(xr0, kb, g, 1.0f);
    v16bf a1 = load_a_f32(xr1, kb, g, 1.0f);
#pragma unroll
    for (int nt = 0; nt < 4; ++nt) {
      v16bf b = load_b_lds(lw, IN_DIM, nt * 16 + m, kb, g);
      acc0[nt] = WMMA_BF16(a0, b, acc0[nt]);
      acc1[nt] = WMMA_BF16(a1, b, acc1[nt]);
    }
  }
#pragma unroll
  for (int nt = 0; nt < 4; ++nt) {
    int n = nt * 16 + m;
    float bv = bias[n];
#pragma unroll
    for (int v = 0; v < 8; ++v) {
      float v0 = acc0[nt][v] + bv; v0 = v0 > 0.0f ? v0 : 0.0f;
      float v1 = acc1[nt][v] + bv; v1 = v1 > 0.0f ? v1 : 0.0f;
      h[(size_t)(row0 + v + g * 8) * HID + n]      = (__bf16)v0;  // D: row = v + 8g
      h[(size_t)(row0 + 16 + v + g * 8) * HID + n] = (__bf16)v1;
    }
  }
}

// ---- scatter: agg[dst] += h[src]  (one wave per edge, 2 cols/lane) -------
__global__ void k_scatter(const __bf16* __restrict__ h, const int* __restrict__ ei,
                          float* __restrict__ agg, int E) {
  int wave = (int)((blockIdx.x * blockDim.x + threadIdx.x) >> 5);
  int lane = threadIdx.x & 31;
  if (wave >= E) return;
  int src = ei[wave];        // row 0 = src
  int dst = ei[E + wave];    // row 1 = dst
  int c = lane * 2;
  const __bf16* hr = h + (size_t)src * HID;
  float v0 = (float)hr[c];
  float v1 = (float)hr[c + 1];
  float* ar = agg + (size_t)dst * HID;
  atomicAdd(&ar[c], v0);
  atomicAdd(&ar[c + 1], v1);
}

// ---- SAGE layer: h_out = relu((agg*inv_deg) @ Wl + bl + h_in @ Wr) -------
__global__ void k_layer(const float* __restrict__ agg, const float* __restrict__ inv_deg,
                        const __bf16* __restrict__ hin,
                        const float* __restrict__ Wl, const float* __restrict__ bl,
                        const float* __restrict__ Wr,
                        __bf16* __restrict__ hout, int n_waves) {
  __shared__ __bf16 lwl[HID * HID];              // lwl[n*64 + k], 8KB
  __shared__ __bf16 lwr[HID * HID];              // lwr[n*64 + k], 8KB
  for (int t = threadIdx.x; t < HID * HID; t += blockDim.x) {
    int n = t >> 6, k = t & 63;
    lwl[t] = (__bf16)Wl[(size_t)k * HID + n];
    lwr[t] = (__bf16)Wr[(size_t)k * HID + n];
  }
  __syncthreads();

  int wave = (int)((blockIdx.x * blockDim.x + threadIdx.x) >> 5);
  int lane = threadIdx.x & 31;
  if (wave >= n_waves) return;
  int m = lane & 15, g = lane >> 4;
  int row0 = wave * 32;
  float idg0 = inv_deg[row0 + m];
  float idg1 = inv_deg[row0 + 16 + m];
  const float*  ar0 = agg + (size_t)(row0 + m) * HID;
  const float*  ar1 = agg + (size_t)(row0 + 16 + m) * HID;
  const __bf16* hr0 = hin + (size_t)(row0 + m) * HID;
  const __bf16* hr1 = hin + (size_t)(row0 + 16 + m) * HID;

  v8f acc0[4] = {}, acc1[4] = {};
  // (agg * inv_deg) @ Wl
#pragma unroll
  for (int kc = 0; kc < 2; ++kc) {
    int kb = kc * 32;
    v16bf a0 = load_a_f32(ar0, kb, g, idg0);
    v16bf a1 = load_a_f32(ar1, kb, g, idg1);
#pragma unroll
    for (int nt = 0; nt < 4; ++nt) {
      v16bf b = load_b_lds(lwl, HID, nt * 16 + m, kb, g);
      acc0[nt] = WMMA_BF16(a0, b, acc0[nt]);
      acc1[nt] = WMMA_BF16(a1, b, acc1[nt]);
    }
  }
  // h_in @ Wr (A already bf16 in memory)
#pragma unroll
  for (int kc = 0; kc < 2; ++kc) {
    int kb = kc * 32;
    v16bf a0 = load_a_bf16(hr0, kb, g);
    v16bf a1 = load_a_bf16(hr1, kb, g);
#pragma unroll
    for (int nt = 0; nt < 4; ++nt) {
      v16bf b = load_b_lds(lwr, HID, nt * 16 + m, kb, g);
      acc0[nt] = WMMA_BF16(a0, b, acc0[nt]);
      acc1[nt] = WMMA_BF16(a1, b, acc1[nt]);
    }
  }
#pragma unroll
  for (int nt = 0; nt < 4; ++nt) {
    int n = nt * 16 + m;
    float bv = bl[n];
#pragma unroll
    for (int v = 0; v < 8; ++v) {
      float v0 = acc0[nt][v] + bv; v0 = v0 > 0.0f ? v0 : 0.0f;
      float v1 = acc1[nt][v] + bv; v1 = v1 > 0.0f ? v1 : 0.0f;
      hout[(size_t)(row0 + v + g * 8) * HID + n]      = (__bf16)v0;
      hout[(size_t)(row0 + 16 + v + g * 8) * HID + n] = (__bf16)v1;
    }
  }
}

// ---- predictor: out = relu(concat(h[p0],h[p1]) @ W1 + b1) @ W2 + b2 ------
__global__ void k_pair(const __bf16* __restrict__ h, const int* __restrict__ pairs,
                       const float* __restrict__ W1, const float* __restrict__ b1,
                       const float* __restrict__ W2, const float* __restrict__ b2,
                       float* __restrict__ out, int n_waves) {
  __shared__ __bf16 lw[HID * IN_DIM];            // lw[n*128 + k], 16KB
  for (int t = threadIdx.x; t < HID * IN_DIM; t += blockDim.x) {
    int n = t >> 7, k = t & 127;
    lw[t] = (__bf16)W1[(size_t)k * HID + n];
  }
  __syncthreads();

  int wave = (int)((blockIdx.x * blockDim.x + threadIdx.x) >> 5);
  int lane = threadIdx.x & 31;
  if (wave >= n_waves) return;
  int m = lane & 15, g = lane >> 4;
  int p0 = wave * 32;
  int ia0 = pairs[(size_t)(p0 + m) * 2 + 0];
  int ib0 = pairs[(size_t)(p0 + m) * 2 + 1];
  int ia1 = pairs[(size_t)(p0 + 16 + m) * 2 + 0];
  int ib1 = pairs[(size_t)(p0 + 16 + m) * 2 + 1];
  const __bf16* za0 = h + (size_t)ia0 * HID;
  const __bf16* zb0 = h + (size_t)ib0 * HID;
  const __bf16* za1 = h + (size_t)ia1 * HID;
  const __bf16* zb1 = h + (size_t)ib1 * HID;

  v8f acc0[4] = {}, acc1[4] = {};
#pragma unroll
  for (int kc = 0; kc < 4; ++kc) {               // K = 128 (halves: h[ia] | h[ib])
    int kb = kc * 32;
    const __bf16* z0 = (kc < 2) ? za0 : zb0;
    const __bf16* z1 = (kc < 2) ? za1 : zb1;
    v16bf a0 = load_a_bf16(z0, kb & 63, g);
    v16bf a1 = load_a_bf16(z1, kb & 63, g);
#pragma unroll
    for (int nt = 0; nt < 4; ++nt) {
      v16bf b = load_b_lds(lw, IN_DIM, nt * 16 + m, kb, g);
      acc0[nt] = WMMA_BF16(a0, b, acc0[nt]);
      acc1[nt] = WMMA_BF16(a1, b, acc1[nt]);
    }
  }
  // hidden = relu(acc + b1); per-row dot with W2, butterfly-reduce 16 lanes/half
  float bb = b2[0];
#pragma unroll
  for (int t = 0; t < 2; ++t) {
    v8f* acc = (t == 0) ? acc0 : acc1;
    float part[8];
#pragma unroll
    for (int v = 0; v < 8; ++v) part[v] = 0.0f;
#pragma unroll
    for (int nt = 0; nt < 4; ++nt) {
      int n = nt * 16 + m;
      float bv = b1[n];
      float w2 = W2[n];
#pragma unroll
      for (int v = 0; v < 8; ++v) {
        float hv = acc[nt][v] + bv;
        hv = hv > 0.0f ? hv : 0.0f;
        part[v] += hv * w2;
      }
    }
#pragma unroll
    for (int off = 8; off >= 1; off >>= 1)
#pragma unroll
      for (int v = 0; v < 8; ++v) part[v] += __shfl_xor(part[v], off, 32);
    if (m == 0) {                                // lanes 0 (rows 0-7) and 16 (rows 8-15)
#pragma unroll
      for (int v = 0; v < 8; ++v) out[p0 + t * 16 + g * 8 + v] = part[v] + bb;
    }
  }
}

// ---- launcher ------------------------------------------------------------
extern "C" void kernel_launch(void* const* d_in, const int* in_sizes, int n_in,
                              void* d_out, int out_size, void* d_ws, size_t ws_size,
                              hipStream_t stream) {
  const float* x      = (const float*)d_in[0];
  const int*   ei     = (const int*)  d_in[1];   // [2, NE]
  const int*   pairs  = (const int*)  d_in[2];   // [NP, 2]
  const float* enc_W  = (const float*)d_in[3];
  const float* enc_b  = (const float*)d_in[4];
  const float* Wl     = (const float*)d_in[5];   // [3, 64, 64]
  const float* bl     = (const float*)d_in[6];   // [3, 64]
  const float* Wr     = (const float*)d_in[7];   // [3, 64, 64]
  const float* W1     = (const float*)d_in[8];   // [128, 64]
  const float* b1     = (const float*)d_in[9];
  const float* W2     = (const float*)d_in[10];  // [64, 1]
  const float* b2     = (const float*)d_in[11];
  float* out = (float*)d_out;

  // workspace carve-up (~52 MB): inv_deg f32[N] | agg f32[N,64] | h_a bf16[N,64] | h_b bf16[N,64]
  char* ws = (char*)d_ws;
  size_t off = 0;
  float* inv_deg = (float*)(ws + off); off += (size_t)NN * 4;        off = (off + 255) & ~(size_t)255;
  float* agg     = (float*)(ws + off); off += (size_t)NN * HID * 4;  off = (off + 255) & ~(size_t)255;
  __bf16* h_a    = (__bf16*)(ws + off); off += (size_t)NN * HID * 2; off = (off + 255) & ~(size_t)255;
  __bf16* h_b    = (__bf16*)(ws + off);

  const int gwaves = NN / 32;   // 3125 (exact): 2 tiles of 16 rows per wave
  const int pwaves = NP / 32;   // 6250 (exact)

  // degree -> inverse degree
  k_zero_f32<<<(NN + 255) / 256, 256, 0, stream>>>(inv_deg, NN);
  k_deg<<<(NE + 255) / 256, 256, 0, stream>>>(ei, inv_deg, NE);
  k_invdeg<<<(NN + 255) / 256, 256, 0, stream>>>(inv_deg, NN);

  // encoder
  k_encoder<<<(gwaves + 7) / 8, 256, 0, stream>>>(x, enc_W, enc_b, h_a, gwaves);

  // 3 SAGE layers with double-buffered features
  __bf16* hc = h_a; __bf16* hn = h_b;
  for (int l = 0; l < 3; ++l) {
    k_zero_f32<<<(NN * HID + 255) / 256, 256, 0, stream>>>(agg, NN * HID);
    k_scatter<<<(NE + 7) / 8, 256, 0, stream>>>(hc, ei, agg, NE);
    k_layer<<<(gwaves + 7) / 8, 256, 0, stream>>>(agg, inv_deg, hc,
        Wl + (size_t)l * HID * HID, bl + (size_t)l * HID, Wr + (size_t)l * HID * HID,
        hn, gwaves);
    __bf16* t = hc; hc = hn; hn = t;
  }

  // pair predictor
  k_pair<<<(pwaves + 7) / 8, 256, 0, stream>>>(hc, pairs, W1, b1, W2, b2, out, pwaves);
}
